// _GCNBackbone_33767032881754
// MI455X (gfx1250) — matrix operand order repaired and checked by reading, not compiled, not run on hardware
//
#include <hip/hip_runtime.h>

// ---------------------------------------------------------------------------
// GCN backbone for MI455X (gfx1250, wave32, WMMA).
// GEMM: pre-packed bf16 operands, pure b128-load + v_wmma_f32_16x16x32_bf16
// inner loop (fully unrolled via FI template), 16x64 output strip per wave.
// Scatter: global_atomic_add_f32, L2-resident (H+AGG <= 102MB < 192MB L2).
// ---------------------------------------------------------------------------

typedef __attribute__((ext_vector_type(16))) __bf16          v16bf;
typedef __attribute__((ext_vector_type(8)))  float           v8f;
typedef __attribute__((ext_vector_type(8)))  unsigned short  v8u;

#define NN 50000
#define NE 800000
#define NG 64

// round-to-nearest-even f32 -> bf16 bits
__device__ __forceinline__ unsigned short f2bfbits(float f) {
  union { float f; unsigned u; } c; c.f = f;
  unsigned r = c.u + 0x7FFFu + ((c.u >> 16) & 1u);
  return (unsigned short)(r >> 16);
}

__global__ void fill_kernel(float* __restrict__ p, float v, int n) {
  int i = blockIdx.x * blockDim.x + threadIdx.x;
  if (i < n) p[i] = v;
}

// deg[dst] += 1 for every real edge (self loops folded in by init deg=1)
__global__ void deg_kernel(const long long* __restrict__ ei, float* __restrict__ deg, int nE) {
  int i = blockIdx.x * blockDim.x + threadIdx.x;
  if (i < nE) atomicAdd(deg + (int)ei[nE + i], 1.0f);
}

__global__ void dinv_kernel(const float* __restrict__ deg, float* __restrict__ dinv, int n) {
  int i = blockIdx.x * blockDim.x + threadIdx.x;
  if (i < n) { float d = deg[i]; dinv[i] = (d > 0.0f) ? rsqrtf(d) : 0.0f; }
}

// elementwise f32 -> bf16 (used once, for the input features)
__global__ void cvt_bf16_kernel(const float* __restrict__ src, unsigned short* __restrict__ dst,
                                int n) {
  int i = blockIdx.x * blockDim.x + threadIdx.x;
  if (i < n) dst[i] = f2bfbits(src[i]);
}

// W[fi,fo] f32 row-major -> Wt[fo,fi] bf16 (column-major W) so each lane's
// 16 B-fragment elements are two contiguous 16-byte runs.
__global__ void cvt_w_transpose_kernel(const float* __restrict__ W, unsigned short* __restrict__ Wt,
                                       int fi, int fo) {
  int i = blockIdx.x * blockDim.x + threadIdx.x;
  if (i >= fi * fo) return;
  int k = i / fo, c = i - k * fo;
  Wt[(size_t)c * fi + k] = f2bfbits(W[i]);
}

// ---------------------------------------------------------------------------
// H[N,fo] = X[N,FI] @ W[FI,fo]; Xbf row-major bf16, Wt = W^T bf16 [fo,FI].
// One wave32 -> 16 rows x 64 cols (4 accumulators). Fragment layouts per
// CDNA5 ISA 7.12.2: A lane half 0 K{0..7,16..23}, half 1 K{8..15,24..31};
// B lane col N=lane%16, K same as A; C VGPR r -> M = r + 8*(lane/16).
// ---------------------------------------------------------------------------
template <int FI>
__global__ void gemm_wmma_kernel(const unsigned short* __restrict__ Xbf,
                                 const unsigned short* __restrict__ Wt,
                                 float* __restrict__ H, int n, int fo) {
  const int lane   = threadIdx.x & 31;
  const int wave   = blockIdx.x * (blockDim.x >> 5) + (threadIdx.x >> 5);
  const int strips = fo >> 6;                    // 64-wide output strips
  const int tiles  = (n >> 4) * strips;
  if (wave >= tiles) return;                     // wave-uniform: EXEC stays all-ones
  const int tm   = wave / strips;
  const int ts   = wave % strips;
  const int half = lane >> 4;
  const int nrow = lane & 15;

  const unsigned short* xrow = Xbf + (size_t)(tm * 16 + nrow) * FI;
  const unsigned short* wrow = Wt + (size_t)(ts * 64 + nrow) * FI;

  v8f acc[4] = {};
#pragma unroll
  for (int k0 = 0; k0 < FI; k0 += 32) {
    union { v8u h[2]; v16bf v; } A;
    A.h[0] = *(const v8u*)(xrow + k0 + half * 8);        // K kb..kb+7
    A.h[1] = *(const v8u*)(xrow + k0 + half * 8 + 16);   // K kb+16..kb+23
#pragma unroll
    for (int t = 0; t < 4; ++t) {
      const unsigned short* wp = wrow + (size_t)t * 16 * FI + k0 + half * 8;
      union { v8u h[2]; v16bf v; } B;
      B.h[0] = *(const v8u*)(wp);                        // K kb..kb+7
      B.h[1] = *(const v8u*)(wp + 16);                   // K kb+16..kb+23
      acc[t] = __builtin_amdgcn_wmma_f32_16x16x32_bf16(
          /*neg_a=*/false, A.v, /*neg_b=*/false, B.v,
          /*c_mod=*/(short)0, acc[t], /*reuse_a=*/false, /*reuse_b=*/false);
    }
  }

  float* hbase = H + (size_t)tm * 16 * fo + ts * 64 + nrow;
#pragma unroll
  for (int t = 0; t < 4; ++t)
#pragma unroll
    for (int r = 0; r < 8; ++r)
      hbase[(size_t)(r + half * 8) * fo + t * 16] = acc[t][r];
}

// ---------------------------------------------------------------------------
// One block per (edge or self-loop); threads cover the fo-wide feature row
// as float4 -> coalesced fo*4-byte gather per edge + 4 f32 atomics per thread.
// ---------------------------------------------------------------------------
__global__ void scatter_kernel(const float* __restrict__ H, const long long* __restrict__ ei,
                               const float* __restrict__ dinv, float* __restrict__ AGG,
                               int nE, int nN, int fo) {
  const int e = blockIdx.x;
  int s, d;
  if (e < nE) { s = (int)ei[e]; d = (int)ei[nE + e]; }
  else        { s = d = e - nE; }                // self loop
  const float w = dinv[s] * dinv[d];
  const int c0 = threadIdx.x * 4;
  if (c0 >= fo) return;
  const float4 h = *(const float4*)(H + (size_t)s * fo + c0);
  float* out = AGG + (size_t)d * fo + c0;
  atomicAdd(out + 0, h.x * w);
  atomicAdd(out + 1, h.y * w);
  atomicAdd(out + 2, h.z * w);
  atomicAdd(out + 3, h.w * w);
}

// bias + ReLU epilogue; writes f32 (for pooling) AND bf16 (next layer's GEMM
// operand) in one bandwidth-bound pass -- the precision conversion never
// touches the GEMM inner loop.
__global__ void bias_relu_kernel(const float* __restrict__ AGG, const float* __restrict__ b,
                                 float* __restrict__ Xf, unsigned short* __restrict__ Xbf,
                                 int total, int foMask) {
  int i = blockIdx.x * blockDim.x + threadIdx.x;
  if (i >= total) return;
  float v = AGG[i] + b[i & foMask];
  v = v > 0.0f ? v : 0.0f;
  Xf[i]  = v;
  Xbf[i] = f2bfbits(v);
}

// segment_max over graphs; post-ReLU values are >= 0, so signed-int atomicMax
// on the raw float bits is order-preserving with a 0.0f init.
__global__ void pool_kernel(const float* __restrict__ X, const long long* __restrict__ batch,
                            float* __restrict__ out, int n) {
  int i = blockIdx.x * blockDim.x + threadIdx.x;      // over n*128 elements
  if (i >= n * 128) return;
  const int node = i >> 7;
  const int c = i & 127;
  const int g = (int)batch[node];
  atomicMax((int*)(out + (size_t)g * 128 + c), __float_as_int(X[i]));
}

// ---------------------------------------------------------------------------
extern "C" void kernel_launch(void* const* d_in, const int* in_sizes, int n_in,
                              void* d_out, int out_size, void* d_ws, size_t ws_size,
                              hipStream_t stream) {
  (void)in_sizes; (void)n_in; (void)ws_size;
  const float*     x0    = (const float*)d_in[0];
  const long long* ei    = (const long long*)d_in[1];   // int64 [2, E]
  const long long* batch = (const long long*)d_in[2];   // int64 [N]
  const float* Wm[4] = { (const float*)d_in[3], (const float*)d_in[5],
                         (const float*)d_in[7], (const float*)d_in[9] };
  const float* bs[4] = { (const float*)d_in[4], (const float*)d_in[6],
                         (const float*)d_in[8], (const float*)d_in[10] };
  const int fis[4] = {128, 256, 128, 128};
  const int fos[4] = {256, 128, 128, 128};

  // workspace layout: Xf | H | AGG (f32, N*256 each) | deg | dinv | Xbf | Wt
  float* ws   = (float*)d_ws;
  float* Xf   = ws;
  float* Hb   = Xf  + (size_t)NN * 256;
  float* AGG  = Hb  + (size_t)NN * 256;
  float* deg  = AGG + (size_t)NN * 256;
  float* dinv = deg + NN;
  unsigned short* Xbf = (unsigned short*)(dinv + NN);   // N*256 bf16
  unsigned short* Wt  = Xbf + (size_t)NN * 256;         // <= 256*256 bf16

  // symmetric GCN normalization: deg = 1 (self loop) + indegree; dinv = rsqrt
  fill_kernel<<<(NN + 255) / 256, 256, 0, stream>>>(deg, 1.0f, NN);
  deg_kernel<<<(NE + 255) / 256, 256, 0, stream>>>(ei, deg, NE);
  dinv_kernel<<<(NN + 255) / 256, 256, 0, stream>>>(deg, dinv, NN);

  // input features -> bf16 once
  cvt_bf16_kernel<<<((NN * 128) + 255) / 256, 256, 0, stream>>>(x0, Xbf, NN * 128);

  for (int l = 0; l < 4; ++l) {
    const int fi = fis[l], fo = fos[l];

    // pack weights: bf16, transposed
    cvt_w_transpose_kernel<<<((fi * fo) + 255) / 256, 256, 0, stream>>>(Wm[l], Wt, fi, fo);

    // H = X @ W via WMMA (16x64 strip per wave, 4 waves per block)
    const int tiles = (NN / 16) * (fo / 64);
    if (fi == 128)
      gemm_wmma_kernel<128><<<(tiles + 3) / 4, 128, 0, stream>>>(Xbf, Wt, Hb, NN, fo);
    else
      gemm_wmma_kernel<256><<<(tiles + 3) / 4, 128, 0, stream>>>(Xbf, Wt, Hb, NN, fo);

    // normalized scatter-add over edges + self loops
    const int total = NN * fo;
    fill_kernel<<<(total + 255) / 256, 256, 0, stream>>>(AGG, 0.0f, total);
    scatter_kernel<<<NE + NN, fo / 4, 0, stream>>>(Hb, ei, dinv, AGG, NE, NN, fo);

    // epilogue: bias+ReLU -> f32 (pool) and bf16 (next GEMM)
    bias_relu_kernel<<<(total + 255) / 256, 256, 0, stream>>>(AGG, bs[l], Xf, Xbf, total, fo - 1);
  }

  // global max pool over 64 graphs -> d_out [64,128]
  fill_kernel<<<(out_size + 255) / 256, 256, 0, stream>>>((float*)d_out, 0.0f, out_size);
  pool_kernel<<<((NN * 128) + 255) / 256, 256, 0, stream>>>(Xf, batch, (float*)d_out, NN);
}
